// RewardNetwork_87067577025414
// MI455X (gfx1250) — compile-verified, matrix operand order checked by reading, not run on hardware
//
#include <hip/hip_runtime.h>
#include <hip/hip_bf16.h>

typedef __attribute__((ext_vector_type(16))) _Float16 v16h;
typedef __attribute__((ext_vector_type(8)))  _Float16 v8h;
typedef __attribute__((ext_vector_type(8)))  float    v8f;

__device__ __forceinline__ float leakyf(float x) { return x >= 0.f ? x : 0.01f * x; }

// ---------------- degree / dinv ----------------
__global__ void k_fill_f32(float* p, float v, int n) {
    int t = blockIdx.x * blockDim.x + threadIdx.x;
    if (t < n) p[t] = v;
}
__global__ void k_deg_edges(const int* __restrict__ ei, float* __restrict__ deg, int E) {
    int e = blockIdx.x * blockDim.x + threadIdx.x;
    if (e < E) atomicAdd(&deg[ei[E + e]], 1.0f);
}
__global__ void k_rsqrt_inplace(float* p, int n) {
    int t = blockIdx.x * blockDim.x + threadIdx.x;
    if (t < n) p[t] = rsqrtf(p[t]);
}

// ---------------- layer-0 GEMM: (N x 3) @ (3 x 256) ----------------
__global__ void k_gemm0(const float* __restrict__ X, const float* __restrict__ W0,
                        float* __restrict__ OUT, int n) {
    int t = blockIdx.x * blockDim.x + threadIdx.x;
    if (t >= n * 256) return;
    int i = t >> 8, c = t & 255;
    const float* xr = X + (size_t)i * 3;
    OUT[t] = xr[0] * W0[c] + xr[1] * W0[256 + c] + xr[2] * W0[512 + c];
}

// ---------------- W f32 -> f16, pre-swizzled into the WMMA B-register layout --
// Swizzled index: idx = ((ct*8 + kt)*32 + lane)*16 + e
//   lane: hs = lane>>4 selects K-half group, col = ct*16 + (lane&15)
//   e -> K offset: e<8 ? 8*hs + e : 16 + 8*hs + (e-8)   (within 32-wide K tile kt)
__global__ void k_wswz(const float* __restrict__ W, _Float16* __restrict__ S, int n) {
    int idx = blockIdx.x * blockDim.x + threadIdx.x;
    if (idx >= n) return;
    int e  = idx & 15;
    int l  = (idx >> 4) & 31;
    int kt = (idx >> 9) & 7;
    int ct = idx >> 12;
    int hs = l >> 4;
    int col = ct * 16 + (l & 15);
    int k = kt * 32 + ((e < 8) ? (8 * hs + e) : (16 + 8 * hs + (e - 8)));
    S[idx] = (_Float16)W[(size_t)k * 256 + col];
}

// ---------------- WMMA GEMM: OUT[rows x 256] = Hh[rows x 256] @ W (swizzled) ---
// One wave per 16-row tile x 64 cols (4 column tiles register-blocked).
// A (f16, row-major): lane m = lane&15 supplies row, hs = lane>>4;
//   v16h a = { K 8hs..8hs+7 , K 16+8hs..16+8hs+7 } within each 32-wide K tile.
// B: one aligned v16h per (ct,kt) from the pre-swizzled buffer.
// C/D: lane = col (lane&15); VGPR r -> row r + 8*hs.
__global__ __launch_bounds__(32) void k_gemm256_wmma(const _Float16* __restrict__ Hh,
                                                     const _Float16* __restrict__ Wswz,
                                                     float* __restrict__ OUT) {
    const int lane = threadIdx.x;
    const int m  = lane & 15;
    const int hs = lane >> 4;
    const _Float16* arow  = Hh + (size_t)(blockIdx.x * 16 + m) * 256;
    const _Float16* bbase = Wswz + (size_t)blockIdx.y * 16384 + (size_t)lane * 16;

    v8f acc[4] = {};
#pragma unroll
    for (int kt = 0; kt < 8; ++kt) {
        union { v16h v; v8h h[2]; } au;
        au.h[0] = *reinterpret_cast<const v8h*>(arow + kt * 32 + 8 * hs);
        au.h[1] = *reinterpret_cast<const v8h*>(arow + kt * 32 + 16 + 8 * hs);
#pragma unroll
        for (int ct = 0; ct < 4; ++ct) {
            const v16h b = *reinterpret_cast<const v16h*>(bbase + (size_t)(ct * 8 + kt) * 512);
            acc[ct] = __builtin_amdgcn_wmma_f32_16x16x32_f16(false, au.v, false, b,
                                                             (short)0, acc[ct], false, false);
        }
    }
#pragma unroll
    for (int ct = 0; ct < 4; ++ct)
#pragma unroll
        for (int r = 0; r < 8; ++r)
            OUT[(size_t)(blockIdx.x * 16 + r + (hs << 3)) * 256 +
                blockIdx.y * 64 + ct * 16 + m] = acc[ct][r];
}

// ---------------- GCN aggregation ----------------
__global__ void k_agg_init(const float* __restrict__ hw, const float* __restrict__ dinv,
                           float* __restrict__ agg, int n) {
    int t = blockIdx.x * blockDim.x + threadIdx.x;
    if (t >= n * 256) return;
    float s = dinv[t >> 8];
    agg[t] = hw[t] * s * s;
}
__global__ void k_edge_agg(const int* __restrict__ ei, const float* __restrict__ dinv,
                           const float* __restrict__ hw, float* __restrict__ agg, int E) {
    int t = blockIdx.x * blockDim.x + threadIdx.x;
    if (t >= E * 64) return;
    int e = t >> 6;
    int c = (t & 63) << 2;
    int s = ei[e];
    int d = ei[E + e];
    float w = dinv[s] * dinv[d];
    const float4 v = *reinterpret_cast<const float4*>(hw + (size_t)s * 256 + c);
    float* ap = agg + (size_t)d * 256 + c;
    atomicAdd(ap + 0, v.x * w);
    atomicAdd(ap + 1, v.y * w);
    atomicAdd(ap + 2, v.z * w);
    atomicAdd(ap + 3, v.w * w);
}
// combine: out = res ? leaky(agg+bias)+res : agg+bias ; also emit f16 copy for next GEMM
__global__ void k_combine(const float* __restrict__ agg, const float* __restrict__ bias,
                          const float* __restrict__ res, float* __restrict__ out,
                          _Float16* __restrict__ outh, int n) {
    int t = blockIdx.x * blockDim.x + threadIdx.x;
    if (t >= n * 256) return;
    float v = agg[t] + bias[t & 255];
    v = res ? (leakyf(v) + res[t]) : v;
    out[t]  = v;
    outh[t] = (_Float16)v;
}

// ---------------- segment-max pooling (ordered-uint atomicMax) ----------------
__device__ __forceinline__ unsigned enc_f32(float f) {
    unsigned u = __float_as_uint(f);
    return (u & 0x80000000u) ? ~u : (u | 0x80000000u);
}
__device__ __forceinline__ float dec_f32(unsigned u) {
    return __uint_as_float((u & 0x80000000u) ? (u ^ 0x80000000u) : ~u);
}
__global__ void k_pool_init(unsigned* __restrict__ p, int n) {
    int t = blockIdx.x * blockDim.x + threadIdx.x;
    if (t < n) p[t] = 0x007FFFFFu;  // enc(-inf)
}
__global__ void k_pool_max(const float* __restrict__ h, const int* __restrict__ bidx,
                           unsigned* __restrict__ p, int n) {
    int t = blockIdx.x * blockDim.x + threadIdx.x;
    if (t >= n * 256) return;
    int b = bidx[t >> 8];
    atomicMax(&p[(size_t)b * 256 + (t & 255)], enc_f32(h[t]));
}
__global__ void k_pool_decode(const unsigned* __restrict__ p, float* __restrict__ o, int n) {
    int t = blockIdx.x * blockDim.x + threadIdx.x;
    if (t < n) o[t] = dec_f32(p[t]);
}

// ---------------- layernorm (one 256-thread block per row, D <= 256) ----------
__global__ void k_layernorm(const float* __restrict__ in, float* __restrict__ out,
                            const float* __restrict__ g, const float* __restrict__ b, int D) {
    __shared__ float sm[256];
    const int row = blockIdx.x, tid = threadIdx.x;
    float x = (tid < D) ? in[(size_t)row * D + tid] : 0.f;
    sm[tid] = x;
    __syncthreads();
    for (int s = 128; s > 0; s >>= 1) {
        if (tid < s) sm[tid] += sm[tid + s];
        __syncthreads();
    }
    float mean = sm[0] / (float)D;
    __syncthreads();
    float dx = (tid < D) ? (x - mean) : 0.f;
    sm[tid] = dx * dx;
    __syncthreads();
    for (int s = 128; s > 0; s >>= 1) {
        if (tid < s) sm[tid] += sm[tid + s];
        __syncthreads();
    }
    float var = sm[0] / (float)D;
    if (tid < D)
        out[(size_t)row * D + tid] = dx * rsqrtf(var + 1e-5f) * g[tid] + b[tid];
}

// ---------------- small dense layers ----------------
__global__ void k_dense(const float* __restrict__ in, const float* __restrict__ W,
                        const float* __restrict__ bias, float* __restrict__ out,
                        int in_dim, int out_dim, int rows, int do_leaky) {
    int t = blockIdx.x * blockDim.x + threadIdx.x;
    if (t >= rows * out_dim) return;
    int r = t / out_dim, j = t - r * out_dim;
    const float* ir = in + (size_t)r * in_dim;
    float s = bias[j];
    for (int k = 0; k < in_dim; ++k) s += ir[k] * W[(size_t)k * out_dim + j];
    out[t] = do_leaky ? leakyf(s) : s;
}
__global__ void k_mul(const float* __restrict__ a, const float* __restrict__ b,
                      float* __restrict__ o, int n) {
    int t = blockIdx.x * blockDim.x + threadIdx.x;
    if (t < n) o[t] = a[t] * b[t];
}
__global__ void k_final(const float* __restrict__ zn, const float* __restrict__ fc2w,
                        const float* __restrict__ fc2b, float* __restrict__ out, int B) {
    int b = blockIdx.x * blockDim.x + threadIdx.x;
    if (b >= B) return;
    float s = fc2b[0];
    for (int j = 0; j < 64; ++j) s += zn[(size_t)b * 64 + j] * fc2w[j];
    out[b] = s;
}

static inline dim3 g1(long long n) { return dim3((unsigned)((n + 255) / 256)); }

extern "C" void kernel_launch(void* const* d_in, const int* in_sizes, int n_in,
                              void* d_out, int out_size, void* d_ws, size_t ws_size,
                              hipStream_t stream) {
    const float* x    = (const float*)d_in[0];
    const int*   ei   = (const int*)d_in[1];
    const int*   bidx = (const int*)d_in[2];
    const float* a    = (const float*)d_in[3];
    const float* W0 = (const float*)d_in[4],  *b0 = (const float*)d_in[5];
    const float* W1 = (const float*)d_in[6],  *b1 = (const float*)d_in[7];
    const float* W2 = (const float*)d_in[8],  *b2 = (const float*)d_in[9];
    const float* W3 = (const float*)d_in[10], *b3 = (const float*)d_in[11];
    const float* afc1_w = (const float*)d_in[12], *afc1_b = (const float*)d_in[13];
    const float* afc2_w = (const float*)d_in[14], *afc2_b = (const float*)d_in[15];
    const float* afc3_w = (const float*)d_in[16], *afc3_b = (const float*)d_in[17];
    const float* fc1_w  = (const float*)d_in[18], *fc1_b  = (const float*)d_in[19];
    const float* fc2_w  = (const float*)d_in[20], *fc2_b  = (const float*)d_in[21];
    const float* ln1_g = (const float*)d_in[22], *ln1_b = (const float*)d_in[23];
    const float* ln2_g = (const float*)d_in[24], *ln2_b = (const float*)d_in[25];
    const float* ln4_g = (const float*)d_in[26], *ln4_b = (const float*)d_in[27];
    const float* ln5_g = (const float*)d_in[28], *ln5_b = (const float*)d_in[29];

    const int N = in_sizes[0] / 3;
    const int E = in_sizes[1] / 2;
    const int B = in_sizes[3] / 7;
    const int D = 256, HDIM = 56, FC = 64;
    const long long ND = (long long)N * D;

    // ---- workspace carve-up (256B aligned blocks) ----
    char* w = (char*)d_ws;
    auto carve = [&](size_t bytes) {
        void* p = (void*)w;
        w += (bytes + 255) & ~(size_t)255;
        return p;
    };
    float*    P0   = (float*)carve((size_t)ND * 4);
    float*    P1   = (float*)carve((size_t)ND * 4);
    float*    AGG  = (float*)carve((size_t)ND * 4);
    _Float16* HF16 = (_Float16*)carve((size_t)ND * 2);
    float*    DINV = (float*)carve((size_t)N * 4);
    _Float16* WSWZ = (_Float16*)carve((size_t)3 * 256 * 256 * 2);
    unsigned* PENC = (unsigned*)carve((size_t)B * D * 4);
    float*    VBUF = (float*)carve((size_t)B * D * 4);
    float*    AV1  = (float*)carve((size_t)B * HDIM * 4);
    float*    AV2  = (float*)carve((size_t)B * HDIM * 4);
    float*    AV3  = (float*)carve((size_t)B * D * 4);
    float*    ZM   = (float*)carve((size_t)B * D * 4);
    float*    ZF   = (float*)carve((size_t)B * FC * 4);

    // ---- pre-swizzle weights into WMMA B-register layout (f16, L2-resident) ----
    k_wswz<<<g1(65536), 256, 0, stream>>>(W1, WSWZ,           65536);
    k_wswz<<<g1(65536), 256, 0, stream>>>(W2, WSWZ + 65536,   65536);
    k_wswz<<<g1(65536), 256, 0, stream>>>(W3, WSWZ + 131072,  65536);

    // ---- degree -> dinv (self-loop counted as 1) ----
    k_fill_f32<<<g1(N), 256, 0, stream>>>(DINV, 1.0f, N);
    k_deg_edges<<<g1(E), 256, 0, stream>>>(ei, DINV, E);
    k_rsqrt_inplace<<<g1(N), 256, 0, stream>>>(DINV, N);

    const dim3 gW(N / 16, 4), bW(32);

    // ---- layer 0: h0 = agg(x@W0) + b0  -> P1 (f32) + HF16 (f16) ----
    k_gemm0<<<g1(ND), 256, 0, stream>>>(x, W0, P1, N);
    k_agg_init<<<g1(ND), 256, 0, stream>>>(P1, DINV, AGG, N);
    k_edge_agg<<<g1((long long)E * 64), 256, 0, stream>>>(ei, DINV, P1, AGG, E);
    k_combine<<<g1(ND), 256, 0, stream>>>(AGG, b0, nullptr, P1, HF16, N);

    // ---- layers 1..3 (ping-pong P1<->P0), hw = h@W via WMMA on f16 A/B ----
    const float* biases[3] = { b1, b2, b3 };
    float* cur = P1;
    float* nxt = P0;
    for (int l = 0; l < 3; ++l) {
        k_gemm256_wmma<<<gW, bW, 0, stream>>>(HF16, WSWZ + (size_t)l * 65536, nxt);
        k_agg_init<<<g1(ND), 256, 0, stream>>>(nxt, DINV, AGG, N);
        k_edge_agg<<<g1((long long)E * 64), 256, 0, stream>>>(ei, DINV, nxt, AGG, E);
        k_combine<<<g1(ND), 256, 0, stream>>>(AGG, biases[l], cur, nxt, HF16, N);
        float* tmp = cur; cur = nxt; nxt = tmp;
    }
    float* H3 = cur;  // final node features (f32)

    // ---- segment-max pool + LN2 ----
    k_pool_init<<<g1((long long)B * D), 256, 0, stream>>>(PENC, B * D);
    k_pool_max<<<g1(ND), 256, 0, stream>>>(H3, bidx, PENC, N);
    k_pool_decode<<<g1((long long)B * D), 256, 0, stream>>>(PENC, VBUF, B * D);
    k_layernorm<<<B, 256, 0, stream>>>(VBUF, VBUF, ln2_g, ln2_b, D);

    // ---- action MLP ----
    k_dense<<<g1((long long)B * HDIM), 256, 0, stream>>>(a, afc1_w, afc1_b, AV1, 7, HDIM, B, 1);
    k_layernorm<<<B, 256, 0, stream>>>(AV1, AV1, ln5_g, ln5_b, HDIM);
    k_dense<<<g1((long long)B * HDIM), 256, 0, stream>>>(AV1, afc2_w, afc2_b, AV2, HDIM, HDIM, B, 1);
    k_dense<<<g1((long long)B * D), 256, 0, stream>>>(AV2, afc3_w, afc3_b, AV3, HDIM, D, B, 1);
    k_layernorm<<<B, 256, 0, stream>>>(AV3, AV3, ln1_g, ln1_b, D);

    // ---- head ----
    k_mul<<<g1((long long)B * D), 256, 0, stream>>>(VBUF, AV3, ZM, B * D);
    k_dense<<<g1((long long)B * FC), 256, 0, stream>>>(ZM, fc1_w, fc1_b, ZF, D, FC, B, 1);
    k_layernorm<<<B, 256, 0, stream>>>(ZF, ZF, ln4_g, ln4_b, FC);
    k_final<<<g1(B), 256, 0, stream>>>(ZF, fc2_w, fc2_b, (float*)d_out, B);
}
